// JointFluxTransformerBlockControl_31155692765818
// MI455X (gfx1250) — matrix operand order repaired and checked by reading, not compiled, not run on hardware
//
#include <hip/hip_runtime.h>
#include <hip/hip_bf16.h>

// ---------------------------------------------------------------------------
// CDNA5 (gfx1250) implementation of a joint Flux transformer block + control.
// All matmuls use v_wmma_f32_16x16x32_bf16 (wave32). f32 weights are converted
// once per block to bf16 through LDS; activations are staged as bf16.
// Attention is flash-style (online softmax) with WMMA for QK^T and PV.
// ---------------------------------------------------------------------------

typedef __attribute__((ext_vector_type(16))) __bf16 v16bf;
typedef __attribute__((ext_vector_type(8)))  float  v8f;

union Frag { v16bf v; unsigned u[8]; };

__device__ inline v8f vzero8() {
  v8f z;
  #pragma unroll
  for (int i = 0; i < 8; ++i) z[i] = 0.f;
  return z;
}

// pack two f32 into one dword of bf16 (low = a, high = b)
__device__ inline unsigned packbf2(float a, float b) {
#if __has_builtin(__builtin_amdgcn_cvt_pk_bf16_f32)
  typedef __attribute__((ext_vector_type(2))) __bf16 v2bf;
  union { v2bf v; unsigned u; } cv;
  cv.v = __builtin_amdgcn_cvt_pk_bf16_f32(a, b);
  return cv.u;
#else
  union { float f; unsigned u; } ca, cb;
  ca.f = a; cb.f = b;
  unsigned ra = ca.u + 0x7FFFu + ((ca.u >> 16) & 1u);
  unsigned rb = cb.u + 0x7FFFu + ((cb.u >> 16) & 1u);
  // v_perm_b32: take high halves of ra (low result) and rb (high result)
  return __builtin_amdgcn_perm(rb, ra, 0x07060302u);
#endif
}

__device__ inline v8f wmma_bf16(const Frag& a, const Frag& b, v8f c) {
  return __builtin_amdgcn_wmma_f32_16x16x32_bf16(
      false, a.v, false, b.v, (short)0, c, false, false);
}

// ---------------------------------------------------------------------------
// GEMM: C[M,N] = A_bf16[M,K] @ W_f32[N,K]^T + bias[N]   (x @ w.T + b)
// Block = 256 threads = 8 waves. Block tile 128(M) x 64(N); wave wv owns rows
// [wv*16, wv*16+16) x all 64 cols = 4 WMMA accumulators. K stepped by 32.
// A tile (128x32 bf16) and W tile (64x32, f32 -> bf16 once) staged in LDS so
// the 8 waves share one weight conversion. M%128==0, N%64==0, K%32==0.
// ---------------------------------------------------------------------------
__global__ __launch_bounds__(256)
void k_gemm_xwT(const __hip_bfloat16* __restrict__ A,
                const float* __restrict__ W,
                const float* __restrict__ bias,
                float* __restrict__ C,
                int M, int N, int K, int ldc)
{
  __shared__ unsigned lds_a[128 * 16];  // 128 rows x 32 bf16
  __shared__ unsigned lds_w[64 * 16];   // 64  rows x 32 bf16

  const int tid  = threadIdx.x;
  const int lane = tid & 31;
  const int wv   = tid >> 5;
  const int hf   = lane >> 4;
  const int l16  = lane & 15;
  const int m0 = blockIdx.y * 128;
  const int n0 = blockIdx.x * 64;
  const unsigned* A32 = (const unsigned*)A;

  const int ar = tid >> 1, as = tid & 1;   // A stage: 128 rows x 2 segs of 8 dwords
  const int wr = tid >> 2, wg = tid & 3;   // W stage: 64 rows x 4 segs of 8 floats

  v8f acc[4];
  #pragma unroll
  for (int t = 0; t < 4; ++t) acc[t] = vzero8();

  for (int k0 = 0; k0 < K; k0 += 32) {
    __syncthreads();
    {  // stage A tile (bf16 passthrough)
      const unsigned* src = A32 + (((size_t)(m0 + ar) * K + k0) >> 1) + as * 8;
      uint4 d0 = *(const uint4*)(src);
      uint4 d1 = *(const uint4*)(src + 4);
      *(uint4*)&lds_a[ar * 16 + as * 8]     = d0;
      *(uint4*)&lds_a[ar * 16 + as * 8 + 4] = d1;
    }
    {  // stage W tile (f32 -> packed bf16, converted once per block)
      const float* src = W + (size_t)(n0 + wr) * K + k0 + wg * 8;
      if (k0 + 32 < K)  // prefetch next chunk of this weight row
        __builtin_prefetch(src + 32, 0, 0);
      float4 f0 = *(const float4*)(src);
      float4 f1 = *(const float4*)(src + 4);
      uint4 p;
      p.x = packbf2(f0.x, f0.y); p.y = packbf2(f0.z, f0.w);
      p.z = packbf2(f1.x, f1.y); p.w = packbf2(f1.z, f1.w);
      *(uint4*)&lds_w[wr * 16 + wg * 4] = p;
    }
    __syncthreads();

    Frag a;  // A fragment: dwords {hf*4..+3, 8+hf*4..+3} of row wv*16+l16
    {
      uint4 lo = *(const uint4*)&lds_a[(wv * 16 + l16) * 16 + hf * 4];
      uint4 hi = *(const uint4*)&lds_a[(wv * 16 + l16) * 16 + 8 + hf * 4];
      a.u[0] = lo.x; a.u[1] = lo.y; a.u[2] = lo.z; a.u[3] = lo.w;
      a.u[4] = hi.x; a.u[5] = hi.y; a.u[6] = hi.z; a.u[7] = hi.w;
    }
    #pragma unroll
    for (int t = 0; t < 4; ++t) {  // B fragment: dwords {hf*4..+3, 8+hf*4..+3} of row t*16+l16
      Frag b;
      uint4 lo = *(const uint4*)&lds_w[(t * 16 + l16) * 16 + hf * 4];
      uint4 hi = *(const uint4*)&lds_w[(t * 16 + l16) * 16 + 8 + hf * 4];
      b.u[0] = lo.x; b.u[1] = lo.y; b.u[2] = lo.z; b.u[3] = lo.w;
      b.u[4] = hi.x; b.u[5] = hi.y; b.u[6] = hi.z; b.u[7] = hi.w;
      acc[t] = wmma_bf16(a, b, acc[t]);
    }
  }
  #pragma unroll
  for (int t = 0; t < 4; ++t) {
    float bs = bias[n0 + 16 * t + l16];
    #pragma unroll
    for (int r = 0; r < 8; ++r) {
      int row = m0 + wv * 16 + r + 8 * hf;
      C[(size_t)row * ldc + n0 + 16 * t + l16] = acc[t][r] + bs;
    }
  }
}

// ---------------------------------------------------------------------------
// Flash attention, one wave per (head, 16-row q tile). Q,K: [H][SA][128] bf16,
// Vt: [H][128][SA] bf16 (transposed for contiguous PV B-fragments).
// Output: [SA][H*128] f32 (merged layout).
// ---------------------------------------------------------------------------
__global__ __launch_bounds__(32)
void k_attn(const __hip_bfloat16* __restrict__ Q,
            const __hip_bfloat16* __restrict__ Km,
            const __hip_bfloat16* __restrict__ Vt,
            float* __restrict__ O, int SA)
{
  __shared__ __hip_bfloat16 lds_p[16 * 32];
  const int lane = threadIdx.x & 31;
  const int hf = lane >> 4, l16 = lane & 15;
  const int hd = blockIdx.y;
  const int q0 = blockIdx.x * 16;
  const unsigned* Q32 = (const unsigned*)Q;
  const unsigned* K32 = (const unsigned*)Km;
  const unsigned* V32 = (const unsigned*)Vt;

  Frag qf[4];
  #pragma unroll
  for (int ds = 0; ds < 4; ++ds)
    #pragma unroll
    for (int v = 0; v < 8; ++v) {
      int g = v >> 2, j = v & 3;
      int d = ds * 32 + g * 16 + hf * 8 + 2 * j;
      qf[ds].u[v] = Q32[((size_t)(hd * SA + q0 + l16) * 128 + d) >> 1];
    }

  v8f o[8];
  #pragma unroll
  for (int dt = 0; dt < 8; ++dt) o[dt] = vzero8();
  float mrow[8], lrow[8];
  #pragma unroll
  for (int r = 0; r < 8; ++r) { mrow[r] = -1e30f; lrow[r] = 0.f; }
  const float scale = 0.08838834764831845f;  // 1/sqrt(128)

  for (int kv0 = 0; kv0 < SA; kv0 += 32) {
    v8f s[2]; s[0] = vzero8(); s[1] = vzero8();
    #pragma unroll
    for (int t = 0; t < 2; ++t)
      #pragma unroll
      for (int ds = 0; ds < 4; ++ds) {
        Frag b;
        #pragma unroll
        for (int v = 0; v < 8; ++v) {
          int d = ds * 32 + (v >> 2) * 16 + hf * 8 + 2 * (v & 3);
          b.u[v] = K32[((size_t)(hd * SA + kv0 + t * 16 + l16) * 128 + d) >> 1];
        }
        s[t] = wmma_bf16(qf[ds], b, s[t]);
      }

    float p0[8], p1[8], alpha[8];
    #pragma unroll
    for (int r = 0; r < 8; ++r) {
      float v0 = s[0][r] * scale, v1 = s[1][r] * scale;
      float mx = fmaxf(v0, v1);
      #pragma unroll
      for (int m = 1; m <= 8; m <<= 1) mx = fmaxf(mx, __shfl_xor(mx, m, 32));
      float mnew = fmaxf(mrow[r], mx);
      alpha[r] = __expf(mrow[r] - mnew);
      p0[r] = __expf(v0 - mnew);
      p1[r] = __expf(v1 - mnew);
      float sum = p0[r] + p1[r];
      #pragma unroll
      for (int m = 1; m <= 8; m <<= 1) sum += __shfl_xor(sum, m, 32);
      lrow[r] = lrow[r] * alpha[r] + sum;
      mrow[r] = mnew;
    }
    #pragma unroll
    for (int dt = 0; dt < 8; ++dt)
      #pragma unroll
      for (int r = 0; r < 8; ++r) o[dt][r] *= alpha[r];

    // C-layout (lane=col) -> A-fragment layout (lane=row) via LDS transpose
    __syncthreads();
    #pragma unroll
    for (int r = 0; r < 8; ++r) {
      lds_p[(r + 8 * hf) * 32 + l16]      = __float2bfloat16(p0[r]);
      lds_p[(r + 8 * hf) * 32 + 16 + l16] = __float2bfloat16(p1[r]);
    }
    __syncthreads();
    Frag pa;
    const unsigned* L32 = (const unsigned*)lds_p;
    #pragma unroll
    for (int v = 0; v < 8; ++v) {
      int g = v >> 2, j = v & 3;
      int kk = g * 16 + hf * 8 + 2 * j;
      pa.u[v] = L32[(l16 * 32 + kk) >> 1];
    }
    #pragma unroll
    for (int dt = 0; dt < 8; ++dt) {
      Frag b;
      #pragma unroll
      for (int v = 0; v < 8; ++v) {
        int kv = kv0 + (v >> 2) * 16 + hf * 8 + 2 * (v & 3);
        b.u[v] = V32[((size_t)(hd * 128 + dt * 16 + l16) * SA + kv) >> 1];
      }
      o[dt] = wmma_bf16(pa, b, o[dt]);
    }
  }
  #pragma unroll
  for (int dt = 0; dt < 8; ++dt)
    #pragma unroll
    for (int r = 0; r < 8; ++r) {
      int row = q0 + r + 8 * hf;
      O[(size_t)row * 1536 + hd * 128 + dt * 16 + l16] = o[dt][r] / lrow[r];
    }
}

// ---------------------------------------------------------------------------
// adaLN projection: out[9216] = silu(temb) @ W^T + b  (tiny GEMV)
// ---------------------------------------------------------------------------
__global__ __launch_bounds__(256)
void k_temb_ada(const float* __restrict__ temb, const float* __restrict__ W,
                const float* __restrict__ b, float* __restrict__ out)
{
  int n = blockIdx.x * 256 + threadIdx.x;
  if (n >= 9216) return;
  const float* wr = W + (size_t)n * 1536;
  float acc = 0.f;
  for (int k = 0; k < 1536; ++k) {
    float t = temb[k];
    acc += (t / (1.f + __expf(-t))) * wr[k];
  }
  out[n] = acc + b[n];
}

// ---------------------------------------------------------------------------
// LayerNorm + adaLN modulation, f32 in -> bf16 out. One block per row.
// ---------------------------------------------------------------------------
__global__ __launch_bounds__(256)
void k_ada_ln(const float* __restrict__ X, const float* __restrict__ shift,
              const float* __restrict__ scale, __hip_bfloat16* __restrict__ Y)
{
  __shared__ float rs[256], rs2[256];
  int row = blockIdx.x;
  const float* xr = X + (size_t)row * 1536;
  float s = 0.f, s2 = 0.f;
  for (int i = threadIdx.x; i < 1536; i += 256) {
    float v = xr[i]; s += v; s2 += v * v;
  }
  rs[threadIdx.x] = s; rs2[threadIdx.x] = s2;
  __syncthreads();
  for (int st = 128; st > 0; st >>= 1) {
    if (threadIdx.x < st) { rs[threadIdx.x] += rs[threadIdx.x + st];
                            rs2[threadIdx.x] += rs2[threadIdx.x + st]; }
    __syncthreads();
  }
  float mu = rs[0] / 1536.f;
  float inv = rsqrtf(rs2[0] / 1536.f - mu * mu + 1e-6f);
  for (int i = threadIdx.x; i < 1536; i += 256) {
    float nx = (xr[i] - mu) * inv;
    Y[(size_t)row * 1536 + i] = __float2bfloat16(nx * (1.f + scale[i]) + shift[i]);
  }
}

// ---------------------------------------------------------------------------
// Per-head RMSNorm + RoPE for q/k: [SA][12][128] f32 -> [12][SA][128] bf16.
// Rows < 512 are the text (added) part -> w_ctx; rows >= 512 -> w_img.
// ---------------------------------------------------------------------------
__global__ __launch_bounds__(128)
void k_qk_prep(const float* __restrict__ X,
               const float* __restrict__ w_ctx, const float* __restrict__ w_img,
               const float* __restrict__ cosr, const float* __restrict__ sinr,
               __hip_bfloat16* __restrict__ Y, int SA)
{
  __shared__ float red[128];
  __shared__ float xn[128];
  int d = threadIdx.x;
  int row = blockIdx.x;
  int hd = blockIdx.y;
  float x = X[(size_t)row * 1536 + hd * 128 + d];
  red[d] = x * x;
  __syncthreads();
  for (int s = 64; s > 0; s >>= 1) {
    if (d < s) red[d] += red[d + s];
    __syncthreads();
  }
  float rms = rsqrtf(red[0] / 128.f + 1e-6f);
  float w = (row < 512) ? w_ctx[d] : w_img[d];
  float v = x * rms * w;
  xn[d] = v;
  __syncthreads();
  float vrot = (d & 1) ? xn[d - 1] : -xn[d + 1];
  float out = v * cosr[(size_t)row * 128 + d] + vrot * sinr[(size_t)row * 128 + d];
  Y[((size_t)hd * SA + row) * 128 + d] = __float2bfloat16(out);
}

// V: [SA][12][128] f32 -> transposed [12][128][SA] bf16
__global__ __launch_bounds__(128)
void k_v_prep(const float* __restrict__ X, __hip_bfloat16* __restrict__ Vt, int SA)
{
  int d = threadIdx.x, row = blockIdx.x, hd = blockIdx.y;
  Vt[((size_t)hd * 128 + d) * SA + row] =
      __float2bfloat16(X[(size_t)row * 1536 + hd * 128 + d]);
}

__global__ void k_add2bf(const float* __restrict__ a, const float* __restrict__ b,
                         __hip_bfloat16* __restrict__ y, int n) {
  int i = blockIdx.x * 256 + threadIdx.x;
  if (i < n) y[i] = __float2bfloat16(a[i] + b[i]);
}
__global__ void k_cvtbf(const float* __restrict__ a, __hip_bfloat16* __restrict__ y, int n) {
  int i = blockIdx.x * 256 + threadIdx.x;
  if (i < n) y[i] = __float2bfloat16(a[i]);
}
// y = a + gate[col] * b   (gate broadcast over rows; may alias y==a)
__global__ void k_gate_add(const float* __restrict__ a, const float* __restrict__ b,
                           const float* __restrict__ gate, float* __restrict__ y, int n) {
  int i = blockIdx.x * 256 + threadIdx.x;
  if (i < n) y[i] = a[i] + gate[i % 1536] * b[i];
}
__global__ void k_gelu_bf(const float* __restrict__ x, __hip_bfloat16* __restrict__ y, int n) {
  int i = blockIdx.x * 256 + threadIdx.x;
  if (i < n) {
    float v = x[i];
    float c = 0.7978845608028654f * (v + 0.044715f * v * v * v);
    y[i] = __float2bfloat16(0.5f * v * (1.f + tanhf(c)));
  }
}

// ---------------------------------------------------------------------------
extern "C" void kernel_launch(void* const* d_in, const int* in_sizes, int n_in,
                              void* d_out, int out_size, void* d_ws, size_t ws_size,
                              hipStream_t stream) {
  (void)in_sizes; (void)n_in; (void)out_size; (void)ws_size;
  const size_t D = 1536, F = 6144, SI = 2048, ST = 512, SA = 2560;

  const float* h_in   = (const float*)d_in[0];
  const float* eh_in  = (const float*)d_in[1];
  const float* temb   = (const float*)d_in[2];
  const float* hc_in  = (const float*)d_in[3];
  const float* ehc_in = (const float*)d_in[4];
  const float* tembc  = (const float*)d_in[5];
  const float* ropec  = (const float*)d_in[6];
  const float* ropes  = (const float*)d_in[7];
#define P(i) ((const float*)d_in[8 + (i)])
  // a_: 0..19  c_: 20..39  norm1*: 40..47  ff/ffc: 48..55

  // ---- workspace layout -------------------------------------------------
  size_t off = 0;
  auto alloc = [&](size_t bytes) { size_t o = off; off += (bytes + 255) & ~(size_t)255; return o; };
  auto F32 = [&](size_t o) { return (float*)((char*)d_ws + o); };
  auto BF  = [&](size_t o) { return (__hip_bfloat16*)((char*)d_ws + o); };

  size_t o_ada  = alloc(4 * 6 * D * 4);
  size_t o_nh   = alloc(SI * D * 2), o_neh = alloc(ST * D * 2);
  size_t o_nhc  = alloc(SI * D * 2), o_nehc = alloc(ST * D * 2);
  size_t o_qA = alloc(SA * D * 4), o_kA = alloc(SA * D * 4), o_vA = alloc(SA * D * 4);
  size_t o_qC = alloc(SA * D * 4), o_kC = alloc(SA * D * 4), o_vC = alloc(SA * D * 4);
  size_t o_Qa = alloc(SA * D * 2), o_Ka = alloc(SA * D * 2), o_Va = alloc(SA * D * 2);
  size_t o_Qc = alloc(SA * D * 2), o_Kc = alloc(SA * D * 2), o_Vc = alloc(SA * D * 2);
  size_t o_at1 = alloc(SA * D * 4), o_atC = alloc(SA * D * 4), o_atX = alloc(SA * D * 4);
  size_t o_obA = alloc(SA * D * 2), o_obC = alloc(SA * D * 2);
  size_t o_aoA = alloc(SA * D * 4), o_aoC = alloc(SA * D * 4);
  size_t o_ln2 = alloc(SI * D * 2);
  size_t o_mid = alloc(SI * F * 4);
  size_t o_mbf = alloc(SI * F * 2);
  size_t o_ffo = alloc(SI * D * 4);

  float* ada_n1 = F32(o_ada);
  float* ada_cx = ada_n1 + 9216;
  float* ada_c  = ada_n1 + 2 * 9216;
  float* ada_cc = ada_n1 + 3 * 9216;

  auto gemm = [&](const __hip_bfloat16* A, const float* W, const float* bias,
                  float* C, int M, int N, int K, int ldc) {
    dim3 g(N / 64, M / 128);
    k_gemm_xwT<<<g, 256, 0, stream>>>(A, W, bias, C, M, N, K, ldc);
  };

  // ---- Phase 0: adaLN vectors ------------------------------------------
  k_temb_ada<<<36, 256, 0, stream>>>(temb,  P(40), P(41), ada_n1);
  k_temb_ada<<<36, 256, 0, stream>>>(temb,  P(42), P(43), ada_cx);
  k_temb_ada<<<36, 256, 0, stream>>>(tembc, P(44), P(45), ada_c);
  k_temb_ada<<<36, 256, 0, stream>>>(tembc, P(46), P(47), ada_cc);

  // ---- Phase 1: LN + modulation -> bf16 --------------------------------
  k_ada_ln<<<(int)SI, 256, 0, stream>>>(h_in,   ada_n1 + 0 * D, ada_n1 + 1 * D, BF(o_nh));
  k_ada_ln<<<(int)ST, 256, 0, stream>>>(eh_in,  ada_cx + 0 * D, ada_cx + 1 * D, BF(o_neh));
  k_ada_ln<<<(int)SI, 256, 0, stream>>>(hc_in,  ada_c  + 0 * D, ada_c  + 1 * D, BF(o_nhc));
  k_ada_ln<<<(int)ST, 256, 0, stream>>>(ehc_in, ada_cc + 0 * D, ada_cc + 1 * D, BF(o_nehc));

  // ---- Phase 2: QKV projections (concat layout: rows 0..511 ctx) -------
  gemm(BF(o_neh), P(6),  P(7),  F32(o_qA),              512, 1536, 1536, 1536);
  gemm(BF(o_nh),  P(0),  P(1),  F32(o_qA) + 512 * D,   2048, 1536, 1536, 1536);
  gemm(BF(o_neh), P(8),  P(9),  F32(o_kA),              512, 1536, 1536, 1536);
  gemm(BF(o_nh),  P(2),  P(3),  F32(o_kA) + 512 * D,   2048, 1536, 1536, 1536);
  gemm(BF(o_neh), P(10), P(11), F32(o_vA),              512, 1536, 1536, 1536);
  gemm(BF(o_nh),  P(4),  P(5),  F32(o_vA) + 512 * D,   2048, 1536, 1536, 1536);
  gemm(BF(o_nehc), P(26), P(27), F32(o_qC),             512, 1536, 1536, 1536);
  gemm(BF(o_nhc),  P(20), P(21), F32(o_qC) + 512 * D,  2048, 1536, 1536, 1536);
  gemm(BF(o_nehc), P(28), P(29), F32(o_kC),             512, 1536, 1536, 1536);
  gemm(BF(o_nhc),  P(22), P(23), F32(o_kC) + 512 * D,  2048, 1536, 1536, 1536);
  gemm(BF(o_nehc), P(30), P(31), F32(o_vC),             512, 1536, 1536, 1536);
  gemm(BF(o_nhc),  P(24), P(25), F32(o_vC) + 512 * D,  2048, 1536, 1536, 1536);

  // ---- Phase 3: RMS + RoPE + head relayout -----------------------------
  dim3 gp((unsigned)SA, 12);
  k_qk_prep<<<gp, 128, 0, stream>>>(F32(o_qA), P(18), P(16), ropec, ropes, BF(o_Qa), (int)SA);
  k_qk_prep<<<gp, 128, 0, stream>>>(F32(o_kA), P(19), P(17), ropec, ropes, BF(o_Ka), (int)SA);
  k_v_prep <<<gp, 128, 0, stream>>>(F32(o_vA), BF(o_Va), (int)SA);
  k_qk_prep<<<gp, 128, 0, stream>>>(F32(o_qC), P(38), P(36), ropec, ropes, BF(o_Qc), (int)SA);
  k_qk_prep<<<gp, 128, 0, stream>>>(F32(o_kC), P(39), P(37), ropec, ropes, BF(o_Kc), (int)SA);
  k_v_prep <<<gp, 128, 0, stream>>>(F32(o_vC), BF(o_Vc), (int)SA);

  // ---- Phase 4: attention x3 (main, control, cross) --------------------
  dim3 ga((unsigned)(SA / 16), 12);
  k_attn<<<ga, 32, 0, stream>>>(BF(o_Qa), BF(o_Ka), BF(o_Va), F32(o_at1), (int)SA);
  k_attn<<<ga, 32, 0, stream>>>(BF(o_Qc), BF(o_Kc), BF(o_Vc), F32(o_atC), (int)SA);
  k_attn<<<ga, 32, 0, stream>>>(BF(o_Qa), BF(o_Kc), BF(o_Vc), F32(o_atX), (int)SA);

  // ---- Phase 5: out = out + out_x; convert to bf16 ---------------------
  int nAll = (int)(SA * D);
  k_add2bf<<<nAll / 256, 256, 0, stream>>>(F32(o_at1), F32(o_atX), BF(o_obA), nAll);
  k_cvtbf <<<nAll / 256, 256, 0, stream>>>(F32(o_atC), BF(o_obC), nAll);

  // ---- Phase 6: output projections -------------------------------------
  gemm(BF(o_obA) + 512 * D, P(12), P(13), F32(o_aoA) + 512 * D, 2048, 1536, 1536, 1536);
  gemm(BF(o_obA),           P(14), P(15), F32(o_aoA),            512, 1536, 1536, 1536);
  gemm(BF(o_obC) + 512 * D, P(32), P(33), F32(o_aoC) + 512 * D, 2048, 1536, 1536, 1536);
  gemm(BF(o_obC),           P(34), P(35), F32(o_aoC),            512, 1536, 1536, 1536);

  // ---- Phase 7: gated attention residuals into d_out -------------------
  float* out_eh  = (float*)d_out;            // [512 x 1536]
  float* out_h   = out_eh + ST * D;          // [2048 x 1536]
  float* out_ehc = out_eh + SA * D;          // [512 x 1536]
  float* out_hc  = out_ehc + ST * D;         // [2048 x 1536]
  int nI = (int)(SI * D), nT = (int)(ST * D);
  k_gate_add<<<nI / 256, 256, 0, stream>>>(h_in,   F32(o_aoA) + 512 * D, ada_n1 + 2 * D, out_h,  nI);
  k_gate_add<<<nT / 256, 256, 0, stream>>>(eh_in,  F32(o_aoA),           ada_cx + 2 * D, out_eh, nT);
  k_gate_add<<<nI / 256, 256, 0, stream>>>(hc_in,  F32(o_aoC) + 512 * D, ada_c  + 2 * D, out_hc, nI);
  k_gate_add<<<nT / 256, 256, 0, stream>>>(ehc_in, F32(o_aoC),           ada_cc + 2 * D, out_ehc, nT);

  // ---- Phase 8: FF blocks (in-place on d_out regions) ------------------
  auto ff = [&](float* X, int S, const float* ada, const float* w1, const float* b1,
                const float* w2, const float* b2) {
    k_ada_ln<<<S, 256, 0, stream>>>(X, ada + 3 * D, ada + 4 * D, BF(o_ln2));
    gemm(BF(o_ln2), w1, b1, F32(o_mid), S, 6144, 1536, 6144);
    int n1 = S * (int)F;
    k_gelu_bf<<<n1 / 256, 256, 0, stream>>>(F32(o_mid), BF(o_mbf), n1);
    gemm(BF(o_mbf), w2, b2, F32(o_ffo), S, 1536, 6144, 1536);
    int n2 = S * (int)D;
    k_gate_add<<<n2 / 256, 256, 0, stream>>>(X, F32(o_ffo), ada + 5 * D, X, n2);
  };
  ff(out_h,   (int)SI, ada_n1, P(48), P(49), P(50), P(51));
  ff(out_eh,  (int)ST, ada_cx, P(52), P(53), P(54), P(55));
  ff(out_hc,  (int)SI, ada_c,  P(48), P(49), P(50), P(51));
  ff(out_ehc, (int)ST, ada_cc, P(52), P(53), P(54), P(55));
#undef P
}
